// MultiHeadAttention_58445914963986
// MI455X (gfx1250) — matrix operand order repaired and checked by reading, not compiled, not run on hardware
//
#include <hip/hip_runtime.h>
#include <hip/hip_bf16.h>

// ---------------------------------------------------------------------------
// MHA forward for MI455X (gfx1250, wave32, WMMA). Fixed shapes hardcoded:
// B=4, N=2048, D=768, H=12, hd=64.
//   0) cvt_f32_bf16            : one-time bf16 conversion of q/k/v + weights
//   1) gemm_bias<bf16 out>     : qw/kw projections (double-buffered LDS)
//   2) gemm_bias_vt            : V projection + transposing epilogue ->
//                                per-head V^T[d][token], token order permuted
//   3) flash_attn              : 64x64 tiles, log2-domain online softmax,
//                                DPP reductions, double-buffered staging
//   4) gemm_bias<f32 out>      : out = attn @ wo^T + bo
// Matmuls via v_wmma_f32_16x16x32_bf16; tile staging via CDNA5
// global_load_async_to_lds_b128 (ASYNCcnt DMA, pre-cast AS3 pointers).
// ---------------------------------------------------------------------------

typedef __bf16 bf16_t;
typedef __attribute__((ext_vector_type(16))) __bf16 v16bf;
typedef __attribute__((ext_vector_type(8)))  __bf16 v8bf;
typedef __attribute__((ext_vector_type(4)))  __bf16 v4bf;
typedef __attribute__((ext_vector_type(8)))  float  v8f;
typedef __attribute__((ext_vector_type(4)))  int    v4i_t;

#define SEQ   2048
#define DM    768
#define BATCH 4
#define HEADS 12
#define HDIM  64
#define KSTEPS (DM / 32)   // 24
#define LOG2E 1.4426950408889634f

#if defined(__has_builtin)
#if __has_builtin(__builtin_amdgcn_global_load_async_to_lds_b128)
#define HAVE_ASYNC_LDS 1
#endif
#endif
#ifndef HAVE_ASYNC_LDS
#define HAVE_ASYNC_LDS 0
#endif

#if HAVE_ASYNC_LDS
typedef __attribute__((address_space(1))) v4i_t* as1_v4i;
typedef __attribute__((address_space(3))) v4i_t* as3_v4i;
typedef __attribute__((address_space(3))) bf16_t* lds_bf16p;
#define LDS_PTR lds_bf16p
__device__ __forceinline__ void cp_async_b128(const bf16_t* g, lds_bf16p l) {
  // (global int4* src, local int4* dst, imm offset, imm cpol)
  __builtin_amdgcn_global_load_async_to_lds_b128((as1_v4i)g, (as3_v4i)l, 0, 0);
}
__device__ __forceinline__ void async_wait_all() {
#if __has_builtin(__builtin_amdgcn_s_wait_asynccnt)
  __builtin_amdgcn_s_wait_asynccnt(0);
#else
  asm volatile("s_wait_asynccnt 0" ::: "memory");
#endif
}
#else
#define LDS_PTR bf16_t*
#endif

// Single-instruction transcendentals (v_exp_f32 is exact exp2 on f32).
__device__ __forceinline__ float fast_exp2(float x) {
#if defined(__has_builtin) && __has_builtin(__builtin_amdgcn_exp2f)
  return __builtin_amdgcn_exp2f(x);
#else
  return exp2f(x);
#endif
}
__device__ __forceinline__ float fast_rcp(float x) {
#if defined(__has_builtin) && __has_builtin(__builtin_amdgcn_rcpf)
  return __builtin_amdgcn_rcpf(x);
#else
  return 1.0f / x;
#endif
}

__device__ __forceinline__ v8f wmma_bf16(v16bf a, v16bf b, v8f c) {
  return __builtin_amdgcn_wmma_f32_16x16x32_bf16(false, a, false, b, (short)0, c,
                                                 false, false);
}

// A-fragment (16x32 MxK): lane = m + 16*g ; elems 0..7 = K 8g.., 8..15 = K 16+8g..
__device__ __forceinline__ v16bf load_frag_a(const bf16_t* base, int ld, int lane) {
  const int g = lane >> 4;
  const int m = lane & 15;
  union { v16bf v; v8bf h[2]; } u;
  u.h[0] = *(const v8bf*)(base + (size_t)m * ld + g * 8);
  u.h[1] = *(const v8bf*)(base + (size_t)m * ld + 16 + g * 8);
  return u.v;
}

// B-fragment (32x16 KxN) from a [n][k] k-contiguous tile: lane n+16g holds
// elems 0..7 = K 8g.., 8..15 = K 16+8g..
__device__ __forceinline__ v16bf load_frag_b(const bf16_t* baseT, int ld, int lane) {
  const int g = lane >> 4;
  const int n = lane & 15;
  const bf16_t* p = baseT + (size_t)n * ld + g * 8;
  union { v16bf v; v8bf h[2]; } u;
  u.h[0] = *(const v8bf*)(p);
  u.h[1] = *(const v8bf*)(p + 16);
  return u.v;
}

// ---- VALU-only 16-lane reductions via DPP ----
template <int CTRL>
__device__ __forceinline__ float dppmov(float x) {
  return __builtin_bit_cast(
      float, __builtin_amdgcn_mov_dpp(__builtin_bit_cast(int, x), CTRL, 0xF, 0xF,
                                      true));
}
__device__ __forceinline__ float red_max16(float v) {
  v = fmaxf(v, dppmov<0xB1>(v));   // quad_perm(1,0,3,2)
  v = fmaxf(v, dppmov<0x4E>(v));   // quad_perm(2,3,0,1)
  v = fmaxf(v, dppmov<0x141>(v));  // row_half_mirror
  v = fmaxf(v, dppmov<0x140>(v));  // row_mirror
  return v;
}
__device__ __forceinline__ float red_sum16(float v) {
  v += dppmov<0xB1>(v);
  v += dppmov<0x4E>(v);
  v += dppmov<0x141>(v);
  v += dppmov<0x140>(v);
  return v;
}

// ---------------------------------------------------------------------------
// One-time f32 -> bf16 conversion, 8 elements/thread.
// ---------------------------------------------------------------------------
__global__ __launch_bounds__(256) void cvt_f32_bf16_kernel(
    const float* __restrict__ s, bf16_t* __restrict__ d) {
  const int i = blockIdx.x * 256 + threadIdx.x;
  const float4* sp = (const float4*)s + (size_t)i * 2;
  float4 a = sp[0], b = sp[1];
  v8bf o;
  o[0] = (__bf16)a.x; o[1] = (__bf16)a.y; o[2] = (__bf16)a.z; o[3] = (__bf16)a.w;
  o[4] = (__bf16)b.x; o[5] = (__bf16)b.y; o[6] = (__bf16)b.z; o[7] = (__bf16)b.w;
  ((v8bf*)d)[i] = o;
}

// ---------------------------------------------------------------------------
// GEMM mainloop (shared by all projection kernels): A(Mx768)@W(768x768)^T,
// both bf16, 128x128 block tile, BK=32, double-buffered LDS, async staging.
// ---------------------------------------------------------------------------
#define GEMM_MAINLOOP(As, Bs, A, W, acc)                                        \
  const int lrow = tid >> 1;                                                    \
  const int lcol = (tid & 1) * 16;                                              \
  const bf16_t* ga = A + (size_t)(bm + lrow) * DM + lcol;                       \
  const bf16_t* gb = W + (size_t)(bn + lrow) * DM + lcol;                       \
  LDS_PTR laddr[2] = {(LDS_PTR)&As[0][lrow * 32 + lcol],                        \
                      (LDS_PTR)&As[1][lrow * 32 + lcol]};                       \
  LDS_PTR lbddr[2] = {(LDS_PTR)&Bs[0][lrow * 32 + lcol],                        \
                      (LDS_PTR)&Bs[1][lrow * 32 + lcol]};                       \
  STAGE_FIRST(ga, gb, laddr[0], lbddr[0]);                                      \
  __syncthreads();                                                              \
  int buf = 0;                                                                  \
  for (int kt = 0; kt < KSTEPS; ++kt) {                                         \
    v8bf ra{}, rb{};                                                            \
    const bool pre = (kt + 1) < KSTEPS;                                         \
    STAGE_BEGIN(pre, ga + (kt + 1) * 32, gb + (kt + 1) * 32, laddr[buf ^ 1],    \
                lbddr[buf ^ 1], ra, rb);                                        \
    v16bf af[4], bfr[2];                                                        \
    _Pragma("unroll") for (int mi = 0; mi < 4; ++mi) af[mi] =                   \
        load_frag_a(&As[buf][(wm * 64 + mi * 16) * 32], 32, lane);              \
    _Pragma("unroll") for (int ni = 0; ni < 2; ++ni) bfr[ni] =                  \
        load_frag_b(&Bs[buf][(wn * 32 + ni * 16) * 32], 32, lane);              \
    _Pragma("unroll") for (int mi = 0; mi < 4; ++mi)                            \
        _Pragma("unroll") for (int ni = 0; ni < 2; ++ni) acc[mi][ni] =          \
            wmma_bf16(af[mi], bfr[ni], acc[mi][ni]);                            \
    STAGE_END(pre, laddr[buf ^ 1], lbddr[buf ^ 1], ra, rb);                     \
    __syncthreads();                                                            \
    buf ^= 1;                                                                   \
  }

#if HAVE_ASYNC_LDS
#define STAGE_FIRST(ga, gb, la, lb)  \
  do { cp_async_b128(ga, la); cp_async_b128(gb, lb); async_wait_all(); } while (0)
#define STAGE_BEGIN(pre, ga, gb, la, lb, ra, rb) \
  do { if (pre) { cp_async_b128(ga, la); cp_async_b128(gb, lb); } } while (0)
#define STAGE_END(pre, la, lb, ra, rb) \
  do { async_wait_all(); } while (0)
#else
#define STAGE_FIRST(ga, gb, la, lb)  \
  do { *(v8bf*)(la) = *(const v8bf*)(ga); *(v8bf*)(lb) = *(const v8bf*)(gb); } while (0)
#define STAGE_BEGIN(pre, ga, gb, la, lb, ra, rb) \
  do { if (pre) { ra = *(const v8bf*)(ga); rb = *(const v8bf*)(gb); } } while (0)
#define STAGE_END(pre, la, lb, ra, rb) \
  do { if (pre) { *(v8bf*)(la) = ra; *(v8bf*)(lb) = rb; } } while (0)
#endif

template <typename TOut>
__global__ __launch_bounds__(256) void gemm_bias_kernel(
    const bf16_t* __restrict__ A, const bf16_t* __restrict__ W,
    const float* __restrict__ bias, TOut* __restrict__ C) {
  __shared__ alignas(16) bf16_t As[2][128 * 32];
  __shared__ alignas(16) bf16_t Bs[2][128 * 32];

  const int tid  = threadIdx.x;
  const int lane = tid & 31;
  const int w    = tid >> 5;
  const int wm   = w >> 2;
  const int wn   = w & 3;
  const int bm   = blockIdx.y * 128;
  const int bn   = blockIdx.x * 128;

  v8f acc[4][2];
#pragma unroll
  for (int mi = 0; mi < 4; ++mi)
#pragma unroll
    for (int ni = 0; ni < 2; ++ni) acc[mi][ni] = (v8f){};

  GEMM_MAINLOOP(As, Bs, A, W, acc)

  const int g  = lane >> 4;
  const int n0 = lane & 15;
#pragma unroll
  for (int ni = 0; ni < 2; ++ni) {
    const int col = bn + wn * 32 + ni * 16 + n0;
    const float bv = bias[col];
#pragma unroll
    for (int mi = 0; mi < 4; ++mi)
#pragma unroll
      for (int r = 0; r < 8; ++r) {
        const int row = bm + wm * 64 + mi * 16 + r + 8 * g;
        C[(size_t)row * DM + col] = (TOut)(acc[mi][ni][r] + bv);
      }
  }
}

// ---------------------------------------------------------------------------
// V projection + transposing epilogue:
//   VT[((b*H + h)*64 + d) * SEQ + tile*64 + c] = Vhead[tile*64 + tok(c)][d],
//   tok(c) = (c&3)*16 + (c>>2)  (matches flash P packed-store layout).
// ---------------------------------------------------------------------------
__global__ __launch_bounds__(256) void gemm_bias_vt_kernel(
    const bf16_t* __restrict__ A, const bf16_t* __restrict__ W,
    const float* __restrict__ bias, bf16_t* __restrict__ VT) {
  __shared__ alignas(16) bf16_t As[2][128 * 32];
  __shared__ alignas(16) bf16_t Bs[2][128 * 32];
  __shared__ alignas(16) bf16_t Ct[128 * 130];

  const int tid  = threadIdx.x;
  const int lane = tid & 31;
  const int w    = tid >> 5;
  const int wm   = w >> 2;
  const int wn   = w & 3;
  const int bm   = blockIdx.y * 128;
  const int bn   = blockIdx.x * 128;

  v8f acc[4][2];
#pragma unroll
  for (int mi = 0; mi < 4; ++mi)
#pragma unroll
    for (int ni = 0; ni < 2; ++ni) acc[mi][ni] = (v8f){};

  GEMM_MAINLOOP(As, Bs, A, W, acc)

  const int g  = lane >> 4;
  const int n0 = lane & 15;
#pragma unroll
  for (int ni = 0; ni < 2; ++ni) {
    const int nl = wn * 32 + ni * 16 + n0;
    const float bv = bias[bn + nl];
#pragma unroll
    for (int mi = 0; mi < 4; ++mi)
#pragma unroll
      for (int r = 0; r < 8; ++r) {
        const int ml = wm * 64 + mi * 16 + r + 8 * g;
        Ct[ml * 130 + nl] = (__bf16)(acc[mi][ni][r] + bv);
      }
  }
  __syncthreads();

  const int b       = bm >> 11;        // bm / SEQ
  const int tokbase = bm & (SEQ - 1);  // bm % SEQ
  const int nidx  = tid >> 4;
  const int chunk = tid & 15;
#pragma unroll
  for (int p = 0; p < 8; ++p) {
    const int nl   = p * 16 + nidx;
    const int ncol = bn + nl;
    const int h    = ncol >> 6;
    const int d    = ncol & 63;
    const int tile = chunk >> 3;
    const int cb   = (chunk & 7) * 8;
    v8bf o;
#pragma unroll
    for (int j = 0; j < 8; ++j) {
      const int c   = cb + j;
      const int tok = ((c & 3) << 4) + (c >> 2) + tile * 64;
      o[j] = Ct[tok * 130 + nl];
    }
    bf16_t* dst = VT + ((size_t)(b * HEADS + h) * HDIM + d) * SEQ + tokbase +
                  tile * 64 + cb;
    *(v8bf*)dst = o;
  }
}

// ---------------------------------------------------------------------------
// Flash attention: grid (SEQ/64, H, B), 128 threads (4 waves); wave = 16 query
// rows; 64-token key tiles, double-buffered staging; 16 WMMAs per tile.
// Online softmax runs in the log2 domain (bare v_exp_f32 per element).
// ---------------------------------------------------------------------------
__global__ __launch_bounds__(128) void flash_attn_kernel(
    const bf16_t* __restrict__ Q, const bf16_t* __restrict__ K,
    const bf16_t* __restrict__ VT, bf16_t* __restrict__ O) {
  __shared__ alignas(16) bf16_t Ks[2][64 * 64];   // [token][dim]
  __shared__ alignas(16) bf16_t Vs[2][64 * 64];   // [dim][c] (permuted tokens)
  __shared__ alignas(16) bf16_t Ps[4][16 * 64];   // per-wave P, [row][c]

  const int b  = blockIdx.z;
  const int h  = blockIdx.y;
  const int qt = blockIdx.x;
  const int tid  = threadIdx.x;
  const int lane = tid & 31;
  const int w    = tid >> 5;
  const int g    = lane >> 4;
  const int n0   = lane & 15;

  const bf16_t* qp  = Q + ((size_t)b * SEQ + qt * 64 + w * 16) * DM + h * HDIM;
  const bf16_t* kp  = K + ((size_t)b * SEQ) * DM + h * HDIM;
  const bf16_t* vtp = VT + (size_t)(b * HEADS + h) * HDIM * SEQ;

  // Q fragments, pre-scaled by 1/sqrt(64)=1/8 (exact in bf16).
  v16bf qf[2];
#pragma unroll
  for (int f = 0; f < 2; ++f) {
    qf[f] = load_frag_a(qp + f * 32, DM, lane);
#pragma unroll
    for (int i = 0; i < 16; ++i)
      qf[f][i] = (__bf16)((float)qf[f][i] * 0.125f);
  }

  v8f o[4];
#pragma unroll
  for (int oi = 0; oi < 4; ++oi) o[oi] = (v8f){};
  float mrow[8], lrow_[8];
#pragma unroll
  for (int r = 0; r < 8; ++r) { mrow[r] = -3.0e38f; lrow_[r] = 0.0f; }

  const int sRow = tid >> 1;
  const int sOff = (tid & 1) * 32;
  const bf16_t* gk0 = kp + (size_t)sRow * DM + sOff;   // += 64*DM per tile
  const bf16_t* gv0 = vtp + (size_t)sRow * SEQ + sOff; // += 64 per tile
  LDS_PTR lk[2] = {(LDS_PTR)&Ks[0][sRow * 64 + sOff],
                   (LDS_PTR)&Ks[1][sRow * 64 + sOff]};
  LDS_PTR lv[2] = {(LDS_PTR)&Vs[0][sRow * 64 + sOff],
                   (LDS_PTR)&Vs[1][sRow * 64 + sOff]};

  // Prologue: stage tile 0.
#if HAVE_ASYNC_LDS
#pragma unroll
  for (int j = 0; j < 2; ++j) {
    cp_async_b128(gk0 + j * 8, lk[0] + j * 8);
    cp_async_b128(gv0 + j * 8, lv[0] + j * 8);
  }
  async_wait_all();
#else
#pragma unroll
  for (int j = 0; j < 2; ++j) {
    *(v8bf*)(lk[0] + j * 8) = *(const v8bf*)(gk0 + j * 8);
    *(v8bf*)(lv[0] + j * 8) = *(const v8bf*)(gv0 + j * 8);
  }
#endif
  __syncthreads();

  int buf = 0;
  for (int kt = 0; kt < SEQ / 64; ++kt) {
    const bool pre = (kt + 1) < SEQ / 64;
    v8bf rk[2]{}, rv[2]{};
#if HAVE_ASYNC_LDS
    if (pre) {
#pragma unroll
      for (int j = 0; j < 2; ++j) {
        cp_async_b128(gk0 + (size_t)(kt + 1) * 64 * DM + j * 8, lk[buf ^ 1] + j * 8);
        cp_async_b128(gv0 + (kt + 1) * 64 + j * 8, lv[buf ^ 1] + j * 8);
      }
    }
#else
    if (pre) {
#pragma unroll
      for (int j = 0; j < 2; ++j) {
        rk[j] = *(const v8bf*)(gk0 + (size_t)(kt + 1) * 64 * DM + j * 8);
        rv[j] = *(const v8bf*)(gv0 + (kt + 1) * 64 + j * 8);
      }
    }
#endif

    // S = (Q/8) @ K^T, then scale into log2 domain (f32, full precision).
    v8f s[4];
#pragma unroll
    for (int nt = 0; nt < 4; ++nt) {
      v16bf kb0 = load_frag_b(&Ks[buf][nt * 16 * 64 + 0],  64, lane);
      v16bf kb1 = load_frag_b(&Ks[buf][nt * 16 * 64 + 32], 64, lane);
      v8f z = (v8f){};
      z = wmma_bf16(qf[0], kb0, z);
      s[nt] = wmma_bf16(qf[1], kb1, z);
#pragma unroll
      for (int r = 0; r < 8; ++r) s[nt][r] *= LOG2E;
    }

    // Online softmax in log2 domain (DPP reductions; single v_exp_f32 each).
    float scale_o[8];
#pragma unroll
    for (int r = 0; r < 8; ++r) {
      float mx = fmaxf(fmaxf(s[0][r], s[1][r]), fmaxf(s[2][r], s[3][r]));
      mx = red_max16(mx);
      const float mnew = fmaxf(mrow[r], mx);
      scale_o[r] = fast_exp2(mrow[r] - mnew);
      s[0][r] = fast_exp2(s[0][r] - mnew);
      s[1][r] = fast_exp2(s[1][r] - mnew);
      s[2][r] = fast_exp2(s[2][r] - mnew);
      s[3][r] = fast_exp2(s[3][r] - mnew);
      const float rs = red_sum16(s[0][r] + s[1][r] + s[2][r] + s[3][r]);
      lrow_[r] = lrow_[r] * scale_o[r] + rs;
      mrow[r] = mnew;
    }
#pragma unroll
    for (int oi = 0; oi < 4; ++oi)
#pragma unroll
      for (int r = 0; r < 8; ++r) o[oi][r] *= scale_o[r];

    // P -> per-wave LDS (one b64 per row per lane; c = n0*4 + nt).
    bf16_t* pw = &Ps[w][0];
#pragma unroll
    for (int r = 0; r < 8; ++r) {
      const int row = r + 8 * g;
      v4bf pv;
      pv[0] = (__bf16)s[0][r];
      pv[1] = (__bf16)s[1][r];
      pv[2] = (__bf16)s[2][r];
      pv[3] = (__bf16)s[3][r];
      *(v4bf*)&pw[row * 64 + n0 * 4] = pv;
    }
    asm volatile("s_wait_dscnt 0" ::: "memory");
    v16bf pf0 = load_frag_a(pw + 0,  64, lane);
    v16bf pf1 = load_frag_a(pw + 32, 64, lane);

    // O += P @ V.
#pragma unroll
    for (int oi = 0; oi < 4; ++oi) {
      v16bf vf0 = load_frag_b(&Vs[buf][oi * 16 * 64 + 0],  64, lane);
      v16bf vf1 = load_frag_b(&Vs[buf][oi * 16 * 64 + 32], 64, lane);
      o[oi] = wmma_bf16(pf0, vf0, o[oi]);
      o[oi] = wmma_bf16(pf1, vf1, o[oi]);
    }

#if HAVE_ASYNC_LDS
    async_wait_all();
#else
    if (pre) {
#pragma unroll
      for (int j = 0; j < 2; ++j) {
        *(v8bf*)(lk[buf ^ 1] + j * 8) = rk[j];
        *(v8bf*)(lv[buf ^ 1] + j * 8) = rv[j];
      }
    }
#endif
    __syncthreads();
    buf ^= 1;
  }

  // Normalize; row-major (B*SEQ, 768) bf16 out for the final projection.
  float inv[8];
#pragma unroll
  for (int r = 0; r < 8; ++r) inv[r] = fast_rcp(lrow_[r]);
  bf16_t* op = O + ((size_t)b * SEQ + qt * 64 + w * 16) * DM + h * HDIM;
#pragma unroll
  for (int oi = 0; oi < 4; ++oi)
#pragma unroll
    for (int r = 0; r < 8; ++r) {
      const int row = r + 8 * g;
      op[(size_t)row * DM + oi * 16 + n0] = (__bf16)(o[oi][r] * inv[r]);
    }
}

// ---------------------------------------------------------------------------
extern "C" void kernel_launch(void* const* d_in, const int* in_sizes, int n_in,
                              void* d_out, int out_size, void* d_ws, size_t ws_size,
                              hipStream_t stream) {
  const float* q  = (const float*)d_in[0];
  const float* k  = (const float*)d_in[1];
  const float* v  = (const float*)d_in[2];
  const float* wq = (const float*)d_in[3];
  const float* bq = (const float*)d_in[4];
  const float* wk = (const float*)d_in[5];
  const float* bk = (const float*)d_in[6];
  const float* wv = (const float*)d_in[7];
  const float* bv = (const float*)d_in[8];
  const float* wo = (const float*)d_in[9];
  const float* bo = (const float*)d_in[10];

  const int M = BATCH * SEQ;          // 8192
  const size_t SZ  = (size_t)M * DM;  // activation elems
  const size_t WSZ = (size_t)DM * DM; // weight elems

  bf16_t* p = (bf16_t*)d_ws;
  bf16_t* qb  = p; p += SZ;
  bf16_t* kb_ = p; p += SZ;
  bf16_t* vb_ = p; p += SZ;
  bf16_t* wqb = p; p += WSZ;
  bf16_t* wkb = p; p += WSZ;
  bf16_t* wvb = p; p += WSZ;
  bf16_t* wob = p; p += WSZ;
  bf16_t* qw  = p; p += SZ;
  bf16_t* kw  = p; p += SZ;
  bf16_t* vt  = p; p += SZ;
  bf16_t* ao  = p; p += SZ;

  // 0) one-time bf16 conversions (8 elems/thread).
  cvt_f32_bf16_kernel<<<(int)(SZ / (256 * 8)), 256, 0, stream>>>(q, qb);
  cvt_f32_bf16_kernel<<<(int)(SZ / (256 * 8)), 256, 0, stream>>>(k, kb_);
  cvt_f32_bf16_kernel<<<(int)(SZ / (256 * 8)), 256, 0, stream>>>(v, vb_);
  cvt_f32_bf16_kernel<<<(int)(WSZ / (256 * 8)), 256, 0, stream>>>(wq, wqb);
  cvt_f32_bf16_kernel<<<(int)(WSZ / (256 * 8)), 256, 0, stream>>>(wk, wkb);
  cvt_f32_bf16_kernel<<<(int)(WSZ / (256 * 8)), 256, 0, stream>>>(wv, wvb);
  cvt_f32_bf16_kernel<<<(int)(WSZ / (256 * 8)), 256, 0, stream>>>(wo, wob);

  dim3 gblk(256);
  dim3 ggrid(DM / 128, M / 128);  // (6, 64)

  gemm_bias_kernel<bf16_t><<<ggrid, gblk, 0, stream>>>(qb, wqb, bq, qw);
  gemm_bias_kernel<bf16_t><<<ggrid, gblk, 0, stream>>>(kb_, wkb, bk, kw);
  gemm_bias_vt_kernel<<<ggrid, gblk, 0, stream>>>(vb_, wvb, bv, vt);

  dim3 agrid(SEQ / 64, HEADS, BATCH);  // (32, 12, 4)
  flash_attn_kernel<<<agrid, dim3(128), 0, stream>>>(qw, kw, vt, ao);

  gemm_bias_kernel<float><<<ggrid, gblk, 0, stream>>>(ao, wob, bo, (float*)d_out);
}
